// HybridQLSTM_65481071406361
// MI455X (gfx1250) — matrix operand order repaired and checked by reading, not compiled, not run on hardware
//
#include <hip/hip_runtime.h>
#include <math.h>

// LSTM: T=512, B=64, D=512, H=512. Persistent-kernel design:
//  - 32 WGs x 256 threads; WG wg owns hidden columns j0..j0+15 (16 cols x 4 gates).
//  - Weight slice packed once into LDS in WMMA B-fragment layout (f16): 128 KB.
//  - combined [64][1024] f16 in LDS: 128 KB, refilled each step (x_t from HBM, h from
//    a global broadcast buffer in d_ws).
//  - Per step: 16 output tiles of 16x16, K=1024 -> v_wmma_f32_16x16x32_f16, 2 tiles/wave.
//  - Elementwise gates + c/h recurrence in LDS, grid-wide barrier per step.

#define T_STEPS 512
#define BATCH   64
#define DIN     512
#define HID     512
#define KTOT    (DIN + HID)   // 1024
#define NWG     32
#define JPW     (HID / NWG)   // 16 hidden columns per workgroup
#define NTHREADS 256

typedef __attribute__((ext_vector_type(16))) _Float16 v16h;
typedef __attribute__((ext_vector_type(8)))  float    v8f;

// LDS layout (bytes):
//  Wfrag : 4 gates * 32 ktiles * 32 lanes * 16 halves = 65536 f16 = 131072
//  Cmb   : 64 * 1024 f16                                        = 131072
//  Gate  : 64 * 64 f32                                          =  16384
//  Cbuf  : 64 * 16 f32                                          =   4096
//  Bias  : 64 f32                                               =    256
#define SMEM_WFRAG 0
#define SMEM_CMB   131072
#define SMEM_GATE  262144
#define SMEM_CBUF  (262144 + 16384)
#define SMEM_BIAS  (262144 + 16384 + 4096)
#define SMEM_BYTES (262144 + 16384 + 4096 + 256)

__global__ void lstm_init_ws(float* hglob, unsigned* cnt) {
    int i = blockIdx.x * blockDim.x + threadIdx.x;
    if (i < BATCH * HID) hglob[i] = 0.0f;   // h_0 = 0
    if (i == 0) *cnt = 0u;                  // barrier counter reset (per call)
}

__device__ __forceinline__ float sigf(float x) { return 1.0f / (1.0f + __expf(-x)); }

extern "C" __global__ __launch_bounds__(NTHREADS)
void lstm_persist(const float* __restrict__ x,
                  const float* __restrict__ Wf, const float* __restrict__ bf,
                  const float* __restrict__ Wi, const float* __restrict__ bi,
                  const float* __restrict__ Wg, const float* __restrict__ bg,
                  const float* __restrict__ Wo, const float* __restrict__ bo,
                  float* __restrict__ out,
                  float* __restrict__ hglob,
                  unsigned* __restrict__ cnt)
{
    extern __shared__ char smem[];
    _Float16* Wfrag  = (_Float16*)(smem + SMEM_WFRAG);
    _Float16* Cmb    = (_Float16*)(smem + SMEM_CMB);
    float*    GateBf = (float*)   (smem + SMEM_GATE);
    float*    Cbuf   = (float*)   (smem + SMEM_CBUF);
    float*    BiasL  = (float*)   (smem + SMEM_BIAS);

    const int tid  = threadIdx.x;
    const int lane = tid & 31;
    const int wv   = tid >> 5;      // wave id 0..7
    const int wg   = blockIdx.x;    // 0..31
    const int j0   = wg * JPW;
    const int hi   = lane >> 4;     // lane half (0/1)
    const int col  = lane & 15;

    const float* Wp[4] = { Wf, Wi, Wg, Wo };
    const float* Bp[4] = { bf, bi, bg, bo };

    // ---- one-time: pack weight slice into B-fragment layout ----
    // Wfrag[((g*32 + kk)*32 + ln)*16 + t] = W_g[ (kk*32 + (ln>>4)*16 + t) * HID + j0 + (ln&15) ]
    for (int e = tid; e < 4 * 32 * 32 * 16; e += NTHREADS) {
        int t16 = e & 15;
        int ln  = (e >> 4) & 31;
        int kk  = (e >> 9) & 31;
        int g   = e >> 14;
        int k   = kk * 32 + (ln >> 4) * 16 + t16;
        int c   = j0 + (ln & 15);
        Wfrag[e] = (_Float16)Wp[g][k * HID + c];
    }
    if (tid < 64) BiasL[tid] = Bp[tid >> 4][j0 + (tid & 15)];
    for (int e = tid; e < BATCH * JPW; e += NTHREADS) Cbuf[e] = 0.0f;
    __syncthreads();

    // tile assignment: wave wv owns tiles tl = 2*wv, 2*wv+1; they share an M-tile
    const int mtile = (wv * 2) >> 2;          // 0..3 (batch rows mtile*16..+15)
    const int g0    = (wv * 2) & 3;           // gate of first tile (0 or 2)
    const int g1    = g0 + 1;
    const int arow  = mtile * 16 + col;       // A-matrix row for this lane

    for (int t = 0; t < T_STEPS; ++t) {
        // ---- fill combined = [x_t | h] as f16 ----
        const float* xt = x + (size_t)t * (BATCH * DIN);
        for (int p = tid; p < (BATCH * DIN) / 2; p += NTHREADS) {
            int m  = p / (DIN / 2);
            int kp = p - m * (DIN / 2);
            float2 v = ((const float2*)xt)[p];
            _Float16* d = Cmb + m * KTOT + kp * 2;
            d[0] = (_Float16)v.x;
            d[1] = (_Float16)v.y;
        }
        for (int p = tid; p < (BATCH * HID) / 2; p += NTHREADS) {
            int m  = p / (HID / 2);
            int kp = p - m * (HID / 2);
            float2 v = ((const float2*)hglob)[p];
            _Float16* d = Cmb + m * KTOT + DIN + kp * 2;
            d[0] = (_Float16)v.x;
            d[1] = (_Float16)v.y;
        }
        if (t + 1 < T_STEPS) {
            // warm L2/WGP$ for the next timestep's activations
            __builtin_prefetch(xt + BATCH * DIN + tid * 64, 0, 1);
        }
        __syncthreads();

        // ---- GEMM tiles: acc += A(16x32,f16) x B(32x16,f16) over K=1024 ----
        v8f acc0 = {};
        v8f acc1 = {};
        for (int kk = 0; kk < 32; ++kk) {
            const int kbase = kk * 32;
            v16h a, b0, b1;
            // A fragment (contiguous k order, matching B packing): halves [kbase+16*hi .. +15]
            const _Float16* pa = Cmb + arow * KTOT + kbase + 16 * hi;
            ((uint4*)&a)[0] = *(const uint4*)(pa);
            ((uint4*)&a)[1] = *(const uint4*)(pa + 8);
            // B fragments: pre-packed contiguous 16 halves per lane
            const _Float16* pb0 = Wfrag + (((g0 * 32 + kk) * 32) + lane) * 16;
            ((uint4*)&b0)[0] = ((const uint4*)pb0)[0];
            ((uint4*)&b0)[1] = ((const uint4*)pb0)[1];
            const _Float16* pb1 = Wfrag + (((g1 * 32 + kk) * 32) + lane) * 16;
            ((uint4*)&b1)[0] = ((const uint4*)pb1)[0];
            ((uint4*)&b1)[1] = ((const uint4*)pb1)[1];
            acc0 = __builtin_amdgcn_wmma_f32_16x16x32_f16(false, a, false, b0,
                                                          (short)0, acc0, false, false);
            acc1 = __builtin_amdgcn_wmma_f32_16x16x32_f16(false, a, false, b1,
                                                          (short)0, acc1, false, false);
        }
        // D layout: lanes 0-15 -> M=v,   N=lane;  lanes 16-31 -> M=v+8, N=lane-16
#pragma unroll
        for (int v = 0; v < 8; ++v) {
            int mrow = mtile * 16 + hi * 8 + v;
            GateBf[mrow * 64 + g0 * 16 + col] = acc0[v];
            GateBf[mrow * 64 + g1 * 16 + col] = acc1[v];
        }
        __syncthreads();

        // ---- elementwise gates + recurrence for our 64x16 h/c slice ----
        for (int e = tid; e < BATCH * JPW; e += NTHREADS) {
            int mm = e >> 4;
            int jj = e & 15;
            const float* gr = GateBf + mm * 64;
            float fg = sigf(gr[jj]        + BiasL[jj]);
            float ig = sigf(gr[16 + jj]   + BiasL[16 + jj]);
            float gg = tanhf(gr[32 + jj]  + BiasL[32 + jj]);
            float og = sigf(gr[48 + jj]   + BiasL[48 + jj]);
            float cv = fg * Cbuf[e] + ig * gg;
            Cbuf[e] = cv;
            float hv = og * tanhf(cv);
            out[(size_t)t * (BATCH * HID) + mm * HID + j0 + jj] = hv;
            hglob[mm * HID + j0 + jj] = hv;
            if (t == T_STEPS - 1) {
                size_t base = (size_t)T_STEPS * (BATCH * HID);
                out[base + mm * HID + j0 + jj] = hv;               // hx
                out[base + BATCH * HID + mm * HID + j0 + jj] = cv; // cx
            }
        }
        __threadfence();
        __syncthreads();

        // ---- grid-wide barrier (monotonic counter; reset each call by init kernel) ----
        if (tid == 0) {
            atomicAdd(cnt, 1u);
            unsigned target = (unsigned)(NWG * (t + 1));
            while (atomicAdd(cnt, 0u) < target) { __builtin_amdgcn_s_sleep(2); }
        }
        __syncthreads();
        __threadfence();   // acquire: other WGs' h stores visible before next fill
    }
}

extern "C" void kernel_launch(void* const* d_in, const int* in_sizes, int n_in,
                              void* d_out, int out_size, void* d_ws, size_t ws_size,
                              hipStream_t stream) {
    (void)in_sizes; (void)n_in; (void)out_size; (void)ws_size;
    const float* x  = (const float*)d_in[0];
    const float* Wf = (const float*)d_in[1];
    const float* bf = (const float*)d_in[2];
    const float* Wi = (const float*)d_in[3];
    const float* bi = (const float*)d_in[4];
    const float* Wg = (const float*)d_in[5];
    const float* bg = (const float*)d_in[6];
    const float* Wo = (const float*)d_in[7];
    const float* bo = (const float*)d_in[8];
    float* out = (float*)d_out;

    float*    hglob = (float*)d_ws;                                  // 64*512 f32
    unsigned* cnt   = (unsigned*)((char*)d_ws + BATCH * HID * 4);    // barrier counter

    lstm_init_ws<<<(BATCH * HID + 255) / 256, 256, 0, stream>>>(hglob, cnt);
    lstm_persist<<<dim3(NWG), dim3(NTHREADS), SMEM_BYTES, stream>>>(
        x, Wf, bf, Wi, bi, Wg, bg, Wo, bo, out, hglob, cnt);
}
